// MultiHeadedSelfAttention_13417477832717
// MI455X (gfx1250) — compile-verified, matrix operand order checked by reading, not run on hardware
//
#include <hip/hip_runtime.h>

// Problem constants (from reference): B=2, S=2048, D=1024, H=16, W=64
#define BB 2
#define SS 2048
#define DD 1024
#define HH 16
#define WW 64
#define BSROWS (BB * SS) /* 4096 */

typedef __bf16 bf16;
typedef __attribute__((ext_vector_type(16))) __bf16 v16bf;
typedef __attribute__((ext_vector_type(8)))  float  v8f;

union FragU { uint4 u[2]; v16bf v; };  // two 16B LDS chunks -> one 16-elem fragment

// ---------------------------------------------------------------------------
// CDNA5 async memory->LDS copy (ASYNCcnt-tracked DMA; ISA 15.18.3 op 98).
// Each active lane copies 16B from its own global address to its own LDS
// offset; completion = data visible in LDS when ASYNCcnt drains.
// ---------------------------------------------------------------------------
__device__ __forceinline__ void async_ld16(void* lds_dst, const void* gsrc) {
  asm volatile("global_load_async_to_lds_b128 %0, %1, off"
               :: "v"((unsigned)(size_t)lds_dst), "v"(gsrc)
               : "memory");
}
#define WAIT_ASYNC(n) asm volatile("s_wait_asynccnt %0" :: "n"(n) : "memory")

// ---------------------------------------------------------------------------
// fp32 -> bf16 cast (4 elems / thread, 8B packed stores)
// ---------------------------------------------------------------------------
__global__ void cast_f32_to_bf16(const float* __restrict__ in,
                                 bf16* __restrict__ out, int n) {
  int i = (blockIdx.x * blockDim.x + threadIdx.x) * 4;
  if (i >= n) return;
  float4 f = *(const float4*)(in + i);
  union { bf16 e[4]; unsigned long long u; } p;
  p.e[0] = (bf16)f.x; p.e[1] = (bf16)f.y; p.e[2] = (bf16)f.z; p.e[3] = (bf16)f.w;
  *(unsigned long long*)(out + i) = p.u;
}

// ---------------------------------------------------------------------------
// fp32 [K,N] -> bf16 transposed [N,K] (one-time per weight; LDS-tiled).
// ---------------------------------------------------------------------------
__global__ void transpose_cast_bf16(const float* __restrict__ in,
                                    bf16* __restrict__ out) {
  __shared__ bf16 tile[64][65];
  const int t = threadIdx.x;
  const int k0 = blockIdx.x * 64;
  const int n0 = blockIdx.y * 64;
  {
    const int kr = t >> 2, off = (t & 3) * 16;
#pragma unroll
    for (int j = 0; j < 4; ++j) {
      float4 f = *(const float4*)(in + (size_t)(k0 + kr) * DD + n0 + off + j * 4);
      tile[kr][off + j * 4 + 0] = (bf16)f.x;
      tile[kr][off + j * 4 + 1] = (bf16)f.y;
      tile[kr][off + j * 4 + 2] = (bf16)f.z;
      tile[kr][off + j * 4 + 3] = (bf16)f.w;
    }
  }
  __syncthreads();
  {
    const int nr = t >> 2, off = (t & 3) * 16;
#pragma unroll
    for (int j = 0; j < 4; ++j) {
      union { bf16 e[4]; unsigned long long u; } p;
#pragma unroll
      for (int ii = 0; ii < 4; ++ii) p.e[ii] = tile[off + j * 4 + ii][nr];
      *(unsigned long long*)(out + (size_t)(n0 + nr) * DD + k0 + off + j * 4) = p.u;
    }
  }
}

// ---------------------------------------------------------------------------
// GEMM: C[M,N] = (A[M,K](bf16) x W(bf16, PRE-TRANSPOSED [N,K]) + bias[N]) * s
// Block 128 thr = 4 waves; block tile 64x64; wave tile 16x64; K chunks of 32.
// Async double-buffered LDS staging; all WMMA operands contiguous b128 loads.
// LAYOUT: 0 = bf16 row-major, 1 = bf16 head-transposed (b,h,w,s), 2 = f32
// ---------------------------------------------------------------------------
template <int LAYOUT>
__global__ void gemm_bf16(const bf16* __restrict__ A, const bf16* __restrict__ Wt,
                          const float* __restrict__ bias, void* __restrict__ outp,
                          int M, int N, int K, float outScale) {
  __shared__ __attribute__((aligned(16))) bf16 As[2][64 * 32];  // [m][k]
  __shared__ __attribute__((aligned(16))) bf16 Bs[2][64 * 32];  // [n][k]

  const int t = threadIdx.x;
  const int wave = t >> 5, lane = t & 31;
  const int lane16 = lane & 15, half = lane >> 4, kbase = half * 8;
  const int m0 = blockIdx.x * 64;
  const int n0 = blockIdx.y * 64;

  v8f zero = {};
  v8f acc[4];
#pragma unroll
  for (int i = 0; i < 4; ++i) acc[i] = zero;

  const int row = t >> 1, off = (t & 1) * 16;  // staging: 64 rows x 32 k
  const bf16* aRow = A  + (size_t)(m0 + row) * K + off;
  const bf16* bRow = Wt + (size_t)(n0 + row) * K + off;

  // 4 async 16B DMAs per thread per K-step (2 for A tile, 2 for B tile)
  auto stage = [&](int kt, int buf) {
    async_ld16(&As[buf][row * 32 + off],     aRow + kt);
    async_ld16(&As[buf][row * 32 + off + 8], aRow + kt + 8);
    async_ld16(&Bs[buf][row * 32 + off],     bRow + kt);
    async_ld16(&Bs[buf][row * 32 + off + 8], bRow + kt + 8);
  };

  stage(0, 0);
  for (int kt = 0; kt < K; kt += 32) {
    const int cur = (kt >> 5) & 1;
    if (kt + 32 < K) {  // overlap next tile's DMA with this tile's WMMAs
      stage(kt + 32, cur ^ 1);
      WAIT_ASYNC(4);    // previous tile's 4 DMAs done; next 4 still in flight
    } else {
      WAIT_ASYNC(0);
    }
    __syncthreads();

    FragU af, bfr[4];
    const bf16* ap = &As[cur][(wave * 16 + lane16) * 32 + kbase];
    af.u[0] = *(const uint4*)ap;
    af.u[1] = *(const uint4*)(ap + 16);
#pragma unroll
    for (int ns = 0; ns < 4; ++ns) {  // preload all B-frags, then WMMA chain
      const bf16* bp = &Bs[cur][(ns * 16 + lane16) * 32 + kbase];
      bfr[ns].u[0] = *(const uint4*)bp;
      bfr[ns].u[1] = *(const uint4*)(bp + 16);
    }
#pragma unroll
    for (int ns = 0; ns < 4; ++ns)
      acc[ns] = __builtin_amdgcn_wmma_f32_16x16x32_bf16(
          false, af.v, false, bfr[ns].v, (short)0, acc[ns], false, false);
    __syncthreads();
  }

  // Epilogue: row = m0 + 16*wave + r + 8*half, col = n0 + 16*ns + lane16
#pragma unroll
  for (int ns = 0; ns < 4; ++ns) {
    const int col = n0 + ns * 16 + lane16;
    const float b = bias[col];
#pragma unroll
    for (int r = 0; r < 8; ++r) {
      const int orow = m0 + wave * 16 + r + 8 * half;
      const float v = (acc[ns][r] + b) * outScale;
      if (LAYOUT == 2) {
        ((float*)outp)[(size_t)orow * N + col] = v;
      } else if (LAYOUT == 1) {
        // (b,h,w,s): b=orow>>11, s=orow&2047, h=col>>6, w=col&63
        const size_t addr =
            ((size_t)((orow >> 11) * HH + (col >> 6)) * WW + (col & 63)) * SS +
            (orow & 2047);
        ((bf16*)outp)[addr] = (bf16)v;
      } else {
        ((bf16*)outp)[(size_t)orow * N + col] = (bf16)v;
      }
    }
  }
}

// ---------------------------------------------------------------------------
// Flash attention (causal). Q,K in (B,S,D) bf16 (Q pre-scaled by 1/8);
// V pre-transposed to (B,H,W,S). Async double-buffered K/V tile staging.
// Block = 128 thr = 4 waves; 64 queries per block, 16 per wave.
// ---------------------------------------------------------------------------
__global__ void flash_attn(const bf16* __restrict__ Q, const bf16* __restrict__ K,
                           const bf16* __restrict__ Vt, bf16* __restrict__ Hout) {
  __shared__ __attribute__((aligned(16))) bf16 Ks[2][64 * 64];   // [key][w]
  __shared__ __attribute__((aligned(16))) bf16 Vs[2][64 * 64];   // [w][key]
  __shared__ __attribute__((aligned(16))) bf16 Ps[4 * 16 * 64];  // per-wave P

  const int t = threadIdx.x;
  const int wave = t >> 5, lane = t & 31;
  const int lane16 = lane & 15, half = lane >> 4, kbase = half * 8;

  const int qTile = blockIdx.x;
  const int bh = blockIdx.y;
  const int b = bh / HH, h = bh % HH;

  const bf16* kRow = K  + (size_t)b * SS * DD + (size_t)h * WW;  // stride DD
  const bf16* vRow = Vt + (size_t)(b * HH + h) * WW * SS;        // stride SS

  // Q fragments (held in registers for the whole pass)
  FragU qf[2];
  {
    const bf16* qp =
        Q + (size_t)b * SS * DD + (size_t)(qTile * 64 + wave * 16 + lane16) * DD +
        (size_t)h * WW;
#pragma unroll
    for (int c = 0; c < 2; ++c) {
      qf[c].u[0] = *(const uint4*)(qp + c * 32 + kbase);
      qf[c].u[1] = *(const uint4*)(qp + c * 32 + kbase + 16);
    }
  }

  v8f zero = {};
  v8f o[4];
#pragma unroll
  for (int i = 0; i < 4; ++i) o[i] = zero;
  float mrow[8], lrow[8];
#pragma unroll
  for (int r = 0; r < 8; ++r) { mrow[r] = -1e30f; lrow[r] = 0.f; }

  const int srow = t >> 1, soff = (t & 1) * 32;  // staging: 64 rows x 64 cols
  // 8 async 16B DMAs per thread per tile (4 for K, 4 for V)
  auto stageKV = [&](int jt, int buf) {
    const bf16* ks = kRow + (size_t)(jt * 64 + srow) * DD + soff;
    const bf16* vs = vRow + (size_t)srow * SS + jt * 64 + soff;
    bf16* kd = &Ks[buf][srow * 64 + soff];
    bf16* vd = &Vs[buf][srow * 64 + soff];
#pragma unroll
    for (int q4 = 0; q4 < 4; ++q4) {
      async_ld16(kd + q4 * 8, ks + q4 * 8);
      async_ld16(vd + q4 * 8, vs + q4 * 8);
    }
  };

  stageKV(0, 0);
  for (int jt = 0; jt <= qTile; ++jt) {
    const int cur = jt & 1;
    if (jt < qTile) {   // next tile's DMA overlaps this tile's compute
      stageKV(jt + 1, cur ^ 1);
      WAIT_ASYNC(8);    // this tile's 8 DMAs done; next tile's 8 in flight
    } else {
      WAIT_ASYNC(0);
    }
    __syncthreads();

    // ---- S = Q x K^T (8 WMMAs) ----
    v8f sc[4];
#pragma unroll
    for (int ns = 0; ns < 4; ++ns) {
      FragU kf0, kf1;
      const bf16* kp = &Ks[cur][(ns * 16 + lane16) * 64];
      kf0.u[0] = *(const uint4*)(kp + kbase);
      kf0.u[1] = *(const uint4*)(kp + kbase + 16);
      kf1.u[0] = *(const uint4*)(kp + 32 + kbase);
      kf1.u[1] = *(const uint4*)(kp + 32 + kbase + 16);
      v8f a = zero;
      a = __builtin_amdgcn_wmma_f32_16x16x32_bf16(false, qf[0].v, false, kf0.v,
                                                  (short)0, a, false, false);
      a = __builtin_amdgcn_wmma_f32_16x16x32_bf16(false, qf[1].v, false, kf1.v,
                                                  (short)0, a, false, false);
      sc[ns] = a;
    }

    // ---- causal mask on diagonal tile ----
    if (jt == qTile) {
#pragma unroll
      for (int ns = 0; ns < 4; ++ns) {
        const int key = ns * 16 + lane16;
#pragma unroll
        for (int r = 0; r < 8; ++r) {
          if (key > wave * 16 + r + 8 * half) sc[ns][r] = -1e30f;
        }
      }
    }

    // ---- online softmax (rows striped across 16 lanes) ----
#pragma unroll
    for (int r = 0; r < 8; ++r) {
      float mx = fmaxf(fmaxf(sc[0][r], sc[1][r]), fmaxf(sc[2][r], sc[3][r]));
#pragma unroll
      for (int m = 8; m >= 1; m >>= 1) mx = fmaxf(mx, __shfl_xor(mx, m, 16));
      const float mnew = fmaxf(mrow[r], mx);
      const float alpha = __expf(mrow[r] - mnew);
      float rs = 0.f;
#pragma unroll
      for (int ns = 0; ns < 4; ++ns) {
        const float p = __expf(sc[ns][r] - mnew);
        sc[ns][r] = p;
        rs += p;
      }
#pragma unroll
      for (int m = 8; m >= 1; m >>= 1) rs += __shfl_xor(rs, m, 16);
      lrow[r] = lrow[r] * alpha + rs;
      mrow[r] = mnew;
#pragma unroll
      for (int v0 = 0; v0 < 4; ++v0) o[v0][r] *= alpha;
    }

    // ---- P (C-layout) -> wave-private LDS -> A-layout ----
    bf16* pw = &Ps[wave * 16 * 64];
#pragma unroll
    for (int ns = 0; ns < 4; ++ns)
#pragma unroll
      for (int r = 0; r < 8; ++r)
        pw[(r + 8 * half) * 64 + ns * 16 + lane16] = (bf16)sc[ns][r];

    // ---- O += P x V (8 WMMAs; V fragments contiguous via [w][key]) ----
#pragma unroll
    for (int c = 0; c < 2; ++c) {
      FragU pf;
      const bf16* pp = &pw[lane16 * 64 + c * 32 + kbase];
      pf.u[0] = *(const uint4*)pp;
      pf.u[1] = *(const uint4*)(pp + 16);
#pragma unroll
      for (int v0 = 0; v0 < 4; ++v0) {
        FragU vf;
        const bf16* vp = &Vs[cur][(v0 * 16 + lane16) * 64 + c * 32 + kbase];
        vf.u[0] = *(const uint4*)vp;
        vf.u[1] = *(const uint4*)(vp + 16);
        o[v0] = __builtin_amdgcn_wmma_f32_16x16x32_bf16(
            false, pf.v, false, vf.v, (short)0, o[v0], false, false);
      }
    }
    __syncthreads();
  }

  // ---- normalize and store H (bf16, merged-head (B,S,D)) ----
#pragma unroll
  for (int v0 = 0; v0 < 4; ++v0) {
#pragma unroll
    for (int r = 0; r < 8; ++r) {
      const int orow = qTile * 64 + wave * 16 + r + 8 * half;
      const int col = h * WW + v0 * 16 + lane16;
      Hout[(size_t)b * SS * DD + (size_t)orow * DD + col] =
          (bf16)(o[v0][r] / lrow[r]);
    }
  }
}

// ---------------------------------------------------------------------------
// Launch: cast/transposes -> QKV GEMMs -> flash attention -> output GEMM
// ws layout (48 MB): xb@0 (8M) | Wt q/k/v/o @8/10/12/14M (2M each)
//                    qb@16M kb@24M vbt@32M hb@40M (8M each)
// ---------------------------------------------------------------------------
extern "C" void kernel_launch(void* const* d_in, const int* in_sizes, int n_in,
                              void* d_out, int out_size, void* d_ws, size_t ws_size,
                              hipStream_t stream) {
  (void)in_sizes; (void)n_in; (void)out_size; (void)ws_size;
  const float* x  = (const float*)d_in[0];
  // d_in[1] = seg (unused by reference math)
  const float* Wq = (const float*)d_in[2];
  const float* bq = (const float*)d_in[3];
  const float* Wk = (const float*)d_in[4];
  const float* bk = (const float*)d_in[5];
  const float* Wv = (const float*)d_in[6];
  const float* bv = (const float*)d_in[7];
  const float* Wo = (const float*)d_in[8];
  const float* bo = (const float*)d_in[9];

  char* ws = (char*)d_ws;
  const size_t MB = 1024u * 1024u;
  bf16* xb  = (bf16*)(ws + 0);
  bf16* wqt = (bf16*)(ws + 8 * MB);
  bf16* wkt = (bf16*)(ws + 10 * MB);
  bf16* wvt = (bf16*)(ws + 12 * MB);
  bf16* wot = (bf16*)(ws + 14 * MB);
  bf16* qb  = (bf16*)(ws + 16 * MB);
  bf16* kb  = (bf16*)(ws + 24 * MB);
  bf16* vbt = (bf16*)(ws + 32 * MB);
  bf16* hb  = (bf16*)(ws + 40 * MB);

  const int nx = BSROWS * DD;  // 4 M elems
  cast_f32_to_bf16<<<(nx / 4 + 255) / 256, 256, 0, stream>>>(x, xb, nx);

  const dim3 tGrid(DD / 64, DD / 64);  // (16,16)
  transpose_cast_bf16<<<tGrid, 256, 0, stream>>>(Wq, wqt);
  transpose_cast_bf16<<<tGrid, 256, 0, stream>>>(Wk, wkt);
  transpose_cast_bf16<<<tGrid, 256, 0, stream>>>(Wv, wvt);
  transpose_cast_bf16<<<tGrid, 256, 0, stream>>>(Wo, wot);

  const dim3 gemmGrid(BSROWS / 64, DD / 64);  // (64, 16)
  const float qScale = 0.125f;                // 1/sqrt(64) folded into Q proj
  gemm_bf16<0><<<gemmGrid, 128, 0, stream>>>(xb, wqt, bq, qb,  BSROWS, DD, DD, qScale);
  gemm_bf16<0><<<gemmGrid, 128, 0, stream>>>(xb, wkt, bk, kb,  BSROWS, DD, DD, 1.0f);
  gemm_bf16<1><<<gemmGrid, 128, 0, stream>>>(xb, wvt, bv, vbt, BSROWS, DD, DD, 1.0f);

  flash_attn<<<dim3(SS / 64, BB * HH), 128, 0, stream>>>(qb, kb, vbt, hb);

  gemm_bf16<2><<<gemmGrid, 128, 0, stream>>>(hb, wot, bo, d_out, BSROWS, DD, DD, 1.0f);
}
